// DynamicConvAttention_48326972014898
// MI455X (gfx1250) — matrix operand order repaired and checked by reading, not compiled, not run on hardware
//
#include <hip/hip_runtime.h>
#include <hip/hip_bf16.h>

typedef __bf16 bf16_t;
typedef __attribute__((ext_vector_type(16))) __bf16 v16bf;
typedef __attribute__((ext_vector_type(8)))  __bf16 v8bf;
typedef __attribute__((ext_vector_type(8)))  float  v8f;

#define BC 4
#define SC 2048
#define DC 1024
#define LN_EPS 1e-6f

// ---------------- WMMA helpers (bf16 16x16x32, f32 acc) ----------------

static __device__ __forceinline__ v8f wmma_bf16(v16bf a, v16bf b, v8f c) {
  // (neg_a, A, neg_b, B, c_mod, C, reuse_a, reuse_b)
  return __builtin_amdgcn_wmma_f32_16x16x32_bf16(false, a, false, b, (short)0, c, false, false);
}

// A-matrix 16x32 bf16 fragment: lane half selects contiguous K (0..15) vs (16..31)
static __device__ __forceinline__ v16bf load_afrag(const bf16_t* arow, int k0, int akb) {
  v16bf f;
  ((v8bf*)&f)[0] = *(const v8bf*)(arow + k0 + 2 * akb);
  ((v8bf*)&f)[1] = *(const v8bf*)(arow + k0 + 2 * akb + 8);
  return f;
}

// B-matrix 32x16 bf16 fragment from B^T row (contiguous 16 K values per lane)
static __device__ __forceinline__ v16bf load_bfrag(const bf16_t* brow, int ks) {
  v16bf f;
  ((v8bf*)&f)[0] = *(const v8bf*)(brow + ks);
  ((v8bf*)&f)[1] = *(const v8bf*)(brow + ks + 8);
  return f;
}

// fp32 row -> bf16 A fragment (for P @ v)
static __device__ __forceinline__ v16bf load_afrag_f32(const float* arow, int k0, int akb) {
  float4 a0 = *(const float4*)(arow + k0 + 2 * akb);
  float4 a1 = *(const float4*)(arow + k0 + 2 * akb + 4);
  float4 a2 = *(const float4*)(arow + k0 + 2 * akb + 8);
  float4 a3 = *(const float4*)(arow + k0 + 2 * akb + 12);
  v16bf f;
  f[0]  = (bf16_t)a0.x; f[1]  = (bf16_t)a0.y; f[2]  = (bf16_t)a0.z; f[3]  = (bf16_t)a0.w;
  f[4]  = (bf16_t)a1.x; f[5]  = (bf16_t)a1.y; f[6]  = (bf16_t)a1.z; f[7]  = (bf16_t)a1.w;
  f[8]  = (bf16_t)a2.x; f[9]  = (bf16_t)a2.y; f[10] = (bf16_t)a2.z; f[11] = (bf16_t)a2.w;
  f[12] = (bf16_t)a3.x; f[13] = (bf16_t)a3.y; f[14] = (bf16_t)a3.z; f[15] = (bf16_t)a3.w;
  return f;
}

// ---------------- Conversion kernels ----------------

__global__ void dca_cvt_x_kernel(const float* __restrict__ x, bf16_t* __restrict__ xb) {
  int i = (blockIdx.x * blockDim.x + threadIdx.x) * 4;
  float4 v = *(const float4*)(x + i);
  xb[i + 0] = (bf16_t)v.x;
  xb[i + 1] = (bf16_t)v.y;
  xb[i + 2] = (bf16_t)v.z;
  xb[i + 3] = (bf16_t)v.w;
}

// dst[n*D + k] = (bf16) src[k*D + n]   (store W^T so B-fragments are contiguous)
__global__ void dca_cvt_wT_kernel(const float* __restrict__ src, bf16_t* __restrict__ dst) {
  int i = blockIdx.x * blockDim.x + threadIdx.x;   // i = n*D + k
  int n = i / DC;
  int k = i % DC;
  dst[i] = (bf16_t)src[(size_t)k * DC + n];
}

// ---------------- Fused QKV GEMM (software-pipelined) ----------------
// q = x@Wq + bq (bf16 row-major), k = x@Wk (bf16 row-major), v = x@Wv + bv (bf16 TRANSPOSED per batch)

__global__ void dca_qkv_kernel(const bf16_t* __restrict__ xb,
                               const bf16_t* __restrict__ wqT,
                               const bf16_t* __restrict__ wkT,
                               const bf16_t* __restrict__ wvT,
                               const float* __restrict__ bq,
                               const float* __restrict__ bv,
                               bf16_t* __restrict__ qb,
                               bf16_t* __restrict__ kout,
                               bf16_t* __restrict__ vT) {
  int wave  = (blockIdx.x * blockDim.x + threadIdx.x) >> 5;
  int lane  = threadIdx.x & 31;
  int tileM = wave / (DC / 16);
  int tileN = wave % (DC / 16);
  int hi    = lane >> 4;
  int lp    = lane & 15;
  int akb   = hi ? 8 : 0;
  int khs   = hi ? 16 : 0;

  const bf16_t* arow = xb + (size_t)(tileM * 16 + lp) * DC;
  const bf16_t* brq  = wqT + (size_t)(tileN * 16 + lp) * DC;
  const bf16_t* brk  = wkT + (size_t)(tileN * 16 + lp) * DC;
  const bf16_t* brv  = wvT + (size_t)(tileN * 16 + lp) * DC;

  v8f accq = {}, acck = {}, accv = {};

  // pipeline prologue: stage k=0 fragments
  v16bf af = load_afrag(arow, 0, akb);
  v16bf fq = load_bfrag(brq, khs);
  v16bf fk = load_bfrag(brk, khs);
  v16bf fv = load_bfrag(brv, khs);

  for (int k0 = 0; k0 < DC; k0 += 32) {
    int k1 = (k0 + 32 < DC) ? (k0 + 32) : 0;        // branchless tail (re-loads k=0)
    __builtin_prefetch(arow + k0 + 256, 0, 0);
    // issue next-iteration loads before this iteration's WMMAs
    v16bf afn = load_afrag(arow, k1, akb);
    v16bf fqn = load_bfrag(brq, k1 + khs);
    v16bf fkn = load_bfrag(brk, k1 + khs);
    v16bf fvn = load_bfrag(brv, k1 + khs);

    accq = wmma_bf16(af, fq, accq);
    acck = wmma_bf16(af, fk, acck);
    accv = wmma_bf16(af, fv, accv);

    af = afn; fq = fqn; fk = fkn; fv = fvn;
  }

  int nn = tileN * 16 + lp;
  float biasq = bq[nn];
  float biasv = bv[nn];
  int mbase = tileM * 16;
  int b = mbase / SC;
#pragma unroll
  for (int r = 0; r < 8; ++r) {
    int m = mbase + r + (hi ? 8 : 0);
    qb[(size_t)m * DC + nn]   = (bf16_t)(accq[r] + biasq);
    kout[(size_t)m * DC + nn] = (bf16_t)acck[r];
    int s = m - b * SC;
    vT[((size_t)b * DC + nn) * SC + s] = (bf16_t)(accv[r] + biasv);
  }
}

// ---------------- Grouped conv1d (K=3, 16 groups) + torch-style LayerNorm ----------------

__global__ void dca_convln_kernel(const float* __restrict__ x,
                                  const float* __restrict__ convw,
                                  const float* __restrict__ convb,
                                  const float* __restrict__ gamma,
                                  const float* __restrict__ beta,
                                  float* __restrict__ convln) {
  int b = blockIdx.x / SC;
  int s = blockIdx.x % SC;
  int tid = threadIdx.x;

  __shared__ float xs[3][DC];
  __shared__ float red[256];

  for (int t = 0; t < 3; ++t) {
    int ss = s + t - 1;
    if (ss >= 0 && ss < SC) {
      const float* src = x + ((size_t)b * SC + ss) * DC;
      for (int i = tid; i < DC; i += 256) xs[t][i] = src[i];
    } else {
      for (int i = tid; i < DC; i += 256) xs[t][i] = 0.0f;
    }
  }
  __syncthreads();

  float y[4];
#pragma unroll
  for (int j = 0; j < 4; ++j) {
    int c = tid + j * 256;
    int gbase = (c >> 6) << 6;
    const float* w = convw + (size_t)c * 64 * 3;
    float acc = convb[c];
    for (int i = 0; i < 64; ++i) {
      int ic = gbase + i;
      acc += xs[0][ic] * w[i * 3 + 0] + xs[1][ic] * w[i * 3 + 1] + xs[2][ic] * w[i * 3 + 2];
    }
    y[j] = acc;
  }

  float part = y[0] + y[1] + y[2] + y[3];
  red[tid] = part;
  __syncthreads();
  for (int off = 128; off > 0; off >>= 1) {
    if (tid < off) red[tid] += red[tid + off];
    __syncthreads();
  }
  float mean = red[0] * (1.0f / (float)DC);
  __syncthreads();

  float q = 0.0f;
#pragma unroll
  for (int j = 0; j < 4; ++j) { float d = y[j] - mean; q += d * d; }
  red[tid] = q;
  __syncthreads();
  for (int off = 128; off > 0; off >>= 1) {
    if (tid < off) red[tid] += red[tid + off];
    __syncthreads();
  }
  float var = red[0] * (1.0f / (float)(DC - 1));   // ddof=1
  float rinv = 1.0f / (sqrtf(var) + LN_EPS);       // eps added to std

  float* dst = convln + ((size_t)b * SC + s) * DC;
#pragma unroll
  for (int j = 0; j < 4; ++j) {
    int c = tid + j * 256;
    dst[c] = gamma[c] * ((y[j] - mean) * rinv) + beta[c];
  }
}

// ---------------- Attention scores: q @ k^T * scale (pipelined) ----------------

__global__ void dca_scores_kernel(const bf16_t* __restrict__ qb,
                                  const bf16_t* __restrict__ kb,
                                  float* __restrict__ scores) {
  const int NSTRIP = SC / 64;                       // 32
  int wave  = (blockIdx.x * blockDim.x + threadIdx.x) >> 5;
  int lane  = threadIdx.x & 31;
  int wpb   = (SC / 16) * NSTRIP;
  int b     = wave / wpb;
  int rem   = wave % wpb;
  int tileM = rem / NSTRIP;
  int strip = rem % NSTRIP;
  int hi    = lane >> 4;
  int lp    = lane & 15;
  int akb   = hi ? 8 : 0;
  int khs   = hi ? 16 : 0;

  const bf16_t* arow = qb + (size_t)(b * SC + tileM * 16 + lp) * DC;
  const bf16_t* brow[4];
#pragma unroll
  for (int t = 0; t < 4; ++t)
    brow[t] = kb + (size_t)(b * SC + strip * 64 + t * 16 + lp) * DC;

  v8f acc[4] = {};
  v16bf af = load_afrag(arow, 0, akb);
  v16bf bfr[4];
#pragma unroll
  for (int t = 0; t < 4; ++t) bfr[t] = load_bfrag(brow[t], khs);

  for (int k0 = 0; k0 < DC; k0 += 32) {
    int k1 = (k0 + 32 < DC) ? (k0 + 32) : 0;
    __builtin_prefetch(arow + k0 + 256, 0, 0);
    v16bf afn = load_afrag(arow, k1, akb);
    v16bf bfn[4];
#pragma unroll
    for (int t = 0; t < 4; ++t) bfn[t] = load_bfrag(brow[t], k1 + khs);

#pragma unroll
    for (int t = 0; t < 4; ++t) acc[t] = wmma_bf16(af, bfr[t], acc[t]);

    af = afn;
#pragma unroll
    for (int t = 0; t < 4; ++t) bfr[t] = bfn[t];
  }

  const float scale = 0.03125f;                     // 1/sqrt(1024)
  size_t base = (size_t)b * SC * SC;
#pragma unroll
  for (int t = 0; t < 4; ++t) {
#pragma unroll
    for (int r = 0; r < 8; ++r) {
      int m = tileM * 16 + r + (hi ? 8 : 0);
      int n = strip * 64 + t * 16 + lp;
      scores[base + (size_t)m * SC + n] = acc[t][r] * scale;
    }
  }
}

// ---------------- Row softmax (in place, fp32) ----------------

__global__ void dca_softmax_kernel(float* __restrict__ scores) {
  float* p = scores + (size_t)blockIdx.x * SC;
  int tid = threadIdx.x;
  __shared__ float red[256];

  float lmax = -3.4e38f;
  for (int i = tid; i < SC; i += 256) lmax = fmaxf(lmax, p[i]);
  red[tid] = lmax;
  __syncthreads();
  for (int off = 128; off > 0; off >>= 1) {
    if (tid < off) red[tid] = fmaxf(red[tid], red[tid + off]);
    __syncthreads();
  }
  float m = red[0];
  __syncthreads();

  float lsum = 0.0f;
  for (int i = tid; i < SC; i += 256) {
    float e = __expf(p[i] - m);
    p[i] = e;
    lsum += e;
  }
  red[tid] = lsum;
  __syncthreads();
  for (int off = 128; off > 0; off >>= 1) {
    if (tid < off) red[tid] += red[tid + off];
    __syncthreads();
  }
  float inv = 1.0f / red[0];
  for (int i = tid; i < SC; i += 256) p[i] *= inv;
}

// ---------------- attn_out = P @ v, accumulated into convln buffer (pipelined) ----------------

__global__ void dca_attnv_kernel(const float* __restrict__ scores,
                                 const bf16_t* __restrict__ vT,
                                 float* __restrict__ convln) {
  const int NSTRIP = DC / 64;                       // 16
  int wave  = (blockIdx.x * blockDim.x + threadIdx.x) >> 5;
  int lane  = threadIdx.x & 31;
  int wpb   = (SC / 16) * NSTRIP;
  int b     = wave / wpb;
  int rem   = wave % wpb;
  int tileM = rem / NSTRIP;
  int strip = rem % NSTRIP;
  int hi    = lane >> 4;
  int lp    = lane & 15;
  int akb   = hi ? 8 : 0;
  int khs   = hi ? 16 : 0;

  const float* arow = scores + (size_t)b * SC * SC + (size_t)(tileM * 16 + lp) * SC;
  const bf16_t* brow[4];
#pragma unroll
  for (int t = 0; t < 4; ++t)
    brow[t] = vT + ((size_t)b * DC + strip * 64 + t * 16 + lp) * SC;

  v8f acc[4] = {};
  v16bf af = load_afrag_f32(arow, 0, akb);
  v16bf bfr[4];
#pragma unroll
  for (int t = 0; t < 4; ++t) bfr[t] = load_bfrag(brow[t], khs);

  for (int k0 = 0; k0 < SC; k0 += 32) {
    int k1 = (k0 + 32 < SC) ? (k0 + 32) : 0;
    __builtin_prefetch(arow + k0 + 128, 0, 0);
    v16bf afn = load_afrag_f32(arow, k1, akb);
    v16bf bfn[4];
#pragma unroll
    for (int t = 0; t < 4; ++t) bfn[t] = load_bfrag(brow[t], k1 + khs);

#pragma unroll
    for (int t = 0; t < 4; ++t) acc[t] = wmma_bf16(af, bfr[t], acc[t]);

    af = afn;
#pragma unroll
    for (int t = 0; t < 4; ++t) bfr[t] = bfn[t];
  }

#pragma unroll
  for (int t = 0; t < 4; ++t) {
#pragma unroll
    for (int r = 0; r < 8; ++r) {
      int m = tileM * 16 + r + (hi ? 8 : 0);
      int n = strip * 64 + t * 16 + lp;
      size_t idx = ((size_t)b * SC + m) * DC + n;
      convln[idx] += acc[t][r];                     // combined_pre = convln + attn_out
    }
  }
}

// ---------------- Second LayerNorm -> bf16 ----------------

__global__ void dca_combln_kernel(const float* __restrict__ pre,
                                  const float* __restrict__ gamma,
                                  const float* __restrict__ beta,
                                  bf16_t* __restrict__ cb) {
  const float* src = pre + (size_t)blockIdx.x * DC;
  bf16_t* dst = cb + (size_t)blockIdx.x * DC;
  int tid = threadIdx.x;
  __shared__ float red[256];

  float y[4];
#pragma unroll
  for (int j = 0; j < 4; ++j) y[j] = src[tid + j * 256];

  float part = y[0] + y[1] + y[2] + y[3];
  red[tid] = part;
  __syncthreads();
  for (int off = 128; off > 0; off >>= 1) {
    if (tid < off) red[tid] += red[tid + off];
    __syncthreads();
  }
  float mean = red[0] * (1.0f / (float)DC);
  __syncthreads();

  float q = 0.0f;
#pragma unroll
  for (int j = 0; j < 4; ++j) { float d = y[j] - mean; q += d * d; }
  red[tid] = q;
  __syncthreads();
  for (int off = 128; off > 0; off >>= 1) {
    if (tid < off) red[tid] += red[tid + off];
    __syncthreads();
  }
  float var = red[0] * (1.0f / (float)(DC - 1));
  float rinv = 1.0f / (sqrtf(var) + LN_EPS);

#pragma unroll
  for (int j = 0; j < 4; ++j) {
    int c = tid + j * 256;
    dst[c] = (bf16_t)(gamma[c] * ((y[j] - mean) * rinv) + beta[c]);
  }
}

// ---------------- Output GEMM: out = combined @ Wo + bo + x (pipelined) ----------------

__global__ void dca_out_kernel(const bf16_t* __restrict__ cb,
                               const bf16_t* __restrict__ woT,
                               const float* __restrict__ bo,
                               const float* __restrict__ x,
                               float* __restrict__ out) {
  const int NSTRIP = DC / 64;                       // 16
  int wave  = (blockIdx.x * blockDim.x + threadIdx.x) >> 5;
  int lane  = threadIdx.x & 31;
  int tileM = wave / NSTRIP;
  int strip = wave % NSTRIP;
  int hi    = lane >> 4;
  int lp    = lane & 15;
  int akb   = hi ? 8 : 0;
  int khs   = hi ? 16 : 0;

  const bf16_t* arow = cb + (size_t)(tileM * 16 + lp) * DC;
  const bf16_t* brow[4];
#pragma unroll
  for (int t = 0; t < 4; ++t)
    brow[t] = woT + (size_t)(strip * 64 + t * 16 + lp) * DC;

  v8f acc[4] = {};
  v16bf af = load_afrag(arow, 0, akb);
  v16bf bfr[4];
#pragma unroll
  for (int t = 0; t < 4; ++t) bfr[t] = load_bfrag(brow[t], khs);

  for (int k0 = 0; k0 < DC; k0 += 32) {
    int k1 = (k0 + 32 < DC) ? (k0 + 32) : 0;
    __builtin_prefetch(arow + k0 + 256, 0, 0);
    v16bf afn = load_afrag(arow, k1, akb);
    v16bf bfn[4];
#pragma unroll
    for (int t = 0; t < 4; ++t) bfn[t] = load_bfrag(brow[t], k1 + khs);

#pragma unroll
    for (int t = 0; t < 4; ++t) acc[t] = wmma_bf16(af, bfr[t], acc[t]);

    af = afn;
#pragma unroll
    for (int t = 0; t < 4; ++t) bfr[t] = bfn[t];
  }

#pragma unroll
  for (int t = 0; t < 4; ++t) {
#pragma unroll
    for (int r = 0; r < 8; ++r) {
      int m = tileM * 16 + r + (hi ? 8 : 0);
      int n = strip * 64 + t * 16 + lp;
      size_t idx = (size_t)m * DC + n;
      out[idx] = acc[t][r] + bo[n] + x[idx];        // + residual
    }
  }
}

// ---------------- Host launcher ----------------

extern "C" void kernel_launch(void* const* d_in, const int* in_sizes, int n_in,
                              void* d_out, int out_size, void* d_ws, size_t ws_size,
                              hipStream_t stream) {
  const float* x     = (const float*)d_in[0];
  const float* Wq    = (const float*)d_in[1];
  const float* bq    = (const float*)d_in[2];
  const float* Wk    = (const float*)d_in[3];
  const float* Wv    = (const float*)d_in[4];
  const float* bv    = (const float*)d_in[5];
  const float* convw = (const float*)d_in[6];
  const float* convb = (const float*)d_in[7];
  const float* gamma = (const float*)d_in[8];
  const float* beta  = (const float*)d_in[9];
  const float* Wo    = (const float*)d_in[10];
  const float* bo    = (const float*)d_in[11];
  float* out = (float*)d_out;

  char* ws = (char*)d_ws;
  const size_t MB = 1024ull * 1024ull;
  bf16_t* xb  = (bf16_t*)(ws + 0);          // 16 MiB : x bf16
  bf16_t* wqT = (bf16_t*)(ws + 16 * MB);    //  2 MiB : Wq^T bf16
  bf16_t* wkT = (bf16_t*)(ws + 18 * MB);
  bf16_t* wvT = (bf16_t*)(ws + 20 * MB);
  bf16_t* woT = (bf16_t*)(ws + 22 * MB);
  bf16_t* qb  = (bf16_t*)(ws + 24 * MB);    // 16 MiB : q bf16
  bf16_t* kb  = (bf16_t*)(ws + 40 * MB);    // 16 MiB : k bf16
  bf16_t* vT  = (bf16_t*)(ws + 56 * MB);    // 16 MiB : v^T bf16
  float*  cl  = (float*)(ws + 72 * MB);     // 32 MiB : convLN, then combined_pre
  float*  sc  = (float*)(ws + 104 * MB);    // 64 MiB : scores / probs fp32
  bf16_t* cb  = (bf16_t*)(ws + 24 * MB);    // reuse q buffer for combined bf16

  const int T = 256;
  const int BSD = BC * SC * DC;             // 8388608

  // 1. conversions
  dca_cvt_x_kernel<<<(BSD / 4) / T, T, 0, stream>>>(x, xb);
  dca_cvt_wT_kernel<<<(DC * DC) / T, T, 0, stream>>>(Wq, wqT);
  dca_cvt_wT_kernel<<<(DC * DC) / T, T, 0, stream>>>(Wk, wkT);
  dca_cvt_wT_kernel<<<(DC * DC) / T, T, 0, stream>>>(Wv, wvT);
  dca_cvt_wT_kernel<<<(DC * DC) / T, T, 0, stream>>>(Wo, woT);

  // 2. fused QKV GEMM: (B*S/16)*(D/16) = 32768 waves -> 4096 blocks
  dca_qkv_kernel<<<(BC * SC / 16) * (DC / 16) / 8, T, 0, stream>>>(
      xb, wqT, wkT, wvT, bq, bv, qb, kb, vT);

  // 3. grouped conv + LN (one block per (b,s) row)
  dca_convln_kernel<<<BC * SC, T, 0, stream>>>(x, convw, convb, gamma, beta, cl);

  // 4. scores: 4 * 128 * 32 = 16384 waves -> 2048 blocks
  dca_scores_kernel<<<BC * (SC / 16) * (SC / 64) / 8, T, 0, stream>>>(qb, kb, sc);

  // 5. softmax: one block per row
  dca_softmax_kernel<<<BC * SC, T, 0, stream>>>(sc);

  // 6. P @ v accumulated into cl: 4 * 128 * 16 = 8192 waves -> 1024 blocks
  dca_attnv_kernel<<<BC * (SC / 16) * (DC / 64) / 8, T, 0, stream>>>(sc, vT, cl);

  // 7. second LN -> bf16
  dca_combln_kernel<<<BC * SC, T, 0, stream>>>(cl, gamma, beta, cb);

  // 8. output GEMM + bias + residual: 512 * 16 = 8192 waves -> 1024 blocks
  dca_out_kernel<<<(BC * SC / 16) * (DC / 64) / 8, T, 0, stream>>>(cb, woT, bo, x, out);
}
